// Transformer_75969381531700
// MI455X (gfx1250) — matrix operand order repaired and checked by reading, not compiled, not run on hardware
//
#include <hip/hip_runtime.h>

// ---------------- problem constants ----------------
#define L_  2
#define B_  8
#define S_  512
#define D_  1024
#define H_  16
#define F_  4096
#define DH_ 64
#define M_  (B_*S_)          // 4096 rows of activations

typedef __bf16 bf16;
typedef __attribute__((ext_vector_type(8)))  bf16  bf16x8;
typedef __attribute__((ext_vector_type(16))) bf16  bf16x16;
typedef __attribute__((ext_vector_type(8)))  float f32x8;

// =====================================================================
// Batched GEMM:  C[b][M,N] = A[b][M,K] (bf16,row) x W[b][N,K]^T (bf16,row) + bias[N]
//
// grid = (N/64, M/128, batch), block = (32,4) = 4 waves.
// Block tile 128x64. Wave w computes a 32x64 register tile:
//   2 M-subtiles x 4 N-subtiles = 8 WMMA accumulators (f32x8 each).
// B tile (64 rows of W x 32 K) is staged in LDS with
// GLOBAL_LOAD_ASYNC_TO_LDS_B128 (ASYNCcnt), double-buffered.
// A fragments are loaded straight from global (b128 pairs) and reused 4x.
// Fragment k order: contiguous k = 16*h + s in both A and B (matched).
// =====================================================================
__device__ __forceinline__ bf16x16 mk16(bf16x8 lo, bf16x8 hi)
{
    return __builtin_shufflevector(lo, hi, 0,1,2,3,4,5,6,7,8,9,10,11,12,13,14,15);
}

__device__ __forceinline__ void async_cp32(unsigned lds_dst, const bf16* src)
{
    unsigned long long a = (unsigned long long)src;
    // INST_OFFSET is added to both the LDS and global address (ISA 10.x async):
    asm volatile("global_load_async_to_lds_b128 %0, %1, off"
                 :: "v"(lds_dst), "v"(a) : "memory");
    asm volatile("global_load_async_to_lds_b128 %0, %1, off offset:16"
                 :: "v"(lds_dst), "v"(a) : "memory");
}

__global__ __launch_bounds__(128)
void k_gemm(float* __restrict__ C, const bf16* __restrict__ A,
            const bf16* __restrict__ W, const float* __restrict__ bias,
            int K, int lda, int ldw, int ldc,
            long long sA, long long sW, long long sC)
{
    // double-buffered B tile: [2][64 n][32 k] bf16 = 8KB
    __shared__ __align__(64) bf16 smB[2][64 * 32];

    const int lane = threadIdx.x;                 // 0..31
    const int wy   = threadIdx.y;                 // 0..3 (wave id)
    const int tn   = blockIdx.x * 64;
    const int tm   = blockIdx.y * 128;
    const long long bz = blockIdx.z;

    const bf16* Ab = A + bz * sA;
    const bf16* Wb = W + bz * sW;
    float*      Cb = C + bz * sC;

    const int sub = lane & 15;                    // A row / B col / C col within 16
    const int h   = lane >> 4;                    // lane-half selector

    // ---- async fill assignment: 128 threads cover 64 n-rows x 64B ----
    const int t  = wy * 32 + lane;                // 0..127
    const int fn = t >> 1;                        // n within tile 0..63
    const int fq = t & 1;                         // which 32B half of the 64B k-run
    const unsigned ldsbase = (unsigned)(unsigned long long)&smB[0][0];
    const bf16* wsrc = Wb + (long long)(tn + fn) * ldw + fq * 16;

    // ---- per-wave A row base (rows tm + wy*32 .. +31) ----
    const int tmw = tm + wy * 32;
    const bf16* pa0 = Ab + (long long)(tmw + sub) * lda + h * 16;       // M-subtile 0
    const bf16* pa1 = pa0 + (long long)16 * lda;                        // M-subtile 1

    f32x8 acc[2][4];
#pragma unroll
    for (int mi = 0; mi < 2; ++mi)
#pragma unroll
        for (int ni = 0; ni < 4; ++ni)
            acc[mi][ni] = f32x8{0.f,0.f,0.f,0.f,0.f,0.f,0.f,0.f};

    // prologue: fill stage 0
    async_cp32(ldsbase + (unsigned)(fn * 32 + fq * 16) * 2, wsrc);

    for (int kk = 0, st = 0; kk < K; kk += 32, ++st) {
        const int cur  = st & 1;
        const bool more = (kk + 32) < K;
        if (more) {
            async_cp32(ldsbase + (unsigned)((cur ^ 1) * 2048 + fn * 32 + fq * 16) * 2,
                       wsrc + (kk + 32));
        }
        // older fill (current stage) completes first: ASYNCcnt is in-order for loads
        if (more) asm volatile("s_wait_asynccnt 0x2" ::: "memory");
        else      asm volatile("s_wait_asynccnt 0x0" ::: "memory");
        __syncthreads();

        // A fragments for this K-step (reused across 4 N-subtiles)
        bf16x16 afr0 = mk16(*(const bf16x8*)(pa0 + kk), *(const bf16x8*)(pa0 + kk + 8));
        bf16x16 afr1 = mk16(*(const bf16x8*)(pa1 + kk), *(const bf16x8*)(pa1 + kk + 8));

        const bf16* smc = &smB[cur][0];
#pragma unroll
        for (int ni = 0; ni < 4; ++ni) {
            // B fragment: col n = ni*16+sub, K run h*16..h*16+15 -> 32B contiguous in LDS
            bf16x16 bfr = *(const bf16x16*)(smc + (ni * 16 + sub) * 32 + h * 16);
            acc[0][ni] = __builtin_amdgcn_wmma_f32_16x16x32_bf16(false, afr0, false, bfr,
                                                                 (short)0, acc[0][ni], false, false);
            acc[1][ni] = __builtin_amdgcn_wmma_f32_16x16x32_bf16(false, afr1, false, bfr,
                                                                 (short)0, acc[1][ni], false, false);
        }
        __syncthreads();   // everyone done reading `cur` before it is refilled
    }

    // ---- epilogue: C[m = h*8 + v][n = sub] per subtile ----
#pragma unroll
    for (int mi = 0; mi < 2; ++mi) {
#pragma unroll
        for (int ni = 0; ni < 4; ++ni) {
            const int cn = tn + ni * 16 + sub;
            const float bv = bias ? bias[cn] : 0.0f;
            float* cp = Cb + (long long)(tmw + mi * 16 + h * 8) * ldc + cn;
#pragma unroll
            for (int v = 0; v < 8; ++v)
                cp[(long long)v * ldc] = acc[mi][ni][v] + bv;
        }
    }
}

// =====================================================================
// Elementwise / reshape kernels
// =====================================================================
__global__ void k_f32_to_bf16(const float* __restrict__ in, bf16* __restrict__ out, int n)
{
    for (int i = blockIdx.x * blockDim.x + threadIdx.x; i < n; i += gridDim.x * blockDim.x)
        out[i] = (bf16)in[i];
}

__global__ void k_copy_both(const float* __restrict__ in, float* __restrict__ of,
                            bf16* __restrict__ ob, int n)
{
    for (int i = blockIdx.x * blockDim.x + threadIdx.x; i < n; i += gridDim.x * blockDim.x) {
        float v = in[i];
        of[i] = v;
        ob[i] = (bf16)v;
    }
}

__global__ void k_relu_bf16(const float* __restrict__ in, bf16* __restrict__ out, int n)
{
    for (int i = blockIdx.x * blockDim.x + threadIdx.x; i < n; i += gridDim.x * blockDim.x)
        out[i] = (bf16)fmaxf(in[i], 0.0f);
}

// (B,S,D) f32 -> (B,H,S,DH) bf16
__global__ void k_split_qk(const float* __restrict__ in, bf16* __restrict__ out)
{
    const int n = M_ * D_;
    for (int i = blockIdx.x * blockDim.x + threadIdx.x; i < n; i += gridDim.x * blockDim.x) {
        int d = i & (DH_ - 1);
        int s = (i >> 6)  & (S_ - 1);
        int h = (i >> 15) & (H_ - 1);
        int b = i >> 19;
        out[i] = (bf16)in[((long long)(b * S_ + s)) * D_ + h * DH_ + d];
    }
}

// (B,S,D) f32 -> (B,H,DH,S) bf16  (transposed V so P@V is A*W^T form)
__global__ void k_split_vt(const float* __restrict__ in, bf16* __restrict__ out)
{
    const int n = M_ * D_;
    for (int i = blockIdx.x * blockDim.x + threadIdx.x; i < n; i += gridDim.x * blockDim.x) {
        int s = i & (S_ - 1);
        int d = (i >> 9)  & (DH_ - 1);
        int h = (i >> 15) & (H_ - 1);
        int b = i >> 19;
        out[i] = (bf16)in[((long long)(b * S_ + s)) * D_ + h * DH_ + d];
    }
}

// (B,H,S,DH) f32 -> (B,S,D) bf16
__global__ void k_merge_heads(const float* __restrict__ in, bf16* __restrict__ out)
{
    const int n = M_ * D_;
    for (int i = blockIdx.x * blockDim.x + threadIdx.x; i < n; i += gridDim.x * blockDim.x) {
        int c = i & (D_ - 1);
        int s = (i >> 10) & (S_ - 1);
        int b = i >> 19;
        int h = c >> 6;
        int d = c & (DH_ - 1);
        out[i] = (bf16)in[(((long long)(b * H_ + h) * S_ + s)) * DH_ + d];
    }
}

// masked (+causal) softmax over rows of length S; scale 1/sqrt(DH)=0.125; writes bf16 P
__global__ __launch_bounds__(128)
void k_softmax(const float* __restrict__ Sc, bf16* __restrict__ P,
               const int* __restrict__ mask, int causal)
{
    const int row = blockIdx.x;              // (b*H + h)*S + q
    const int q   = row & (S_ - 1);
    const int b   = row >> 13;               // / (H_*S_) = /8192
    const int tid = threadIdx.x;
    const long long base = (long long)row * S_;

    float v[4]; float mx = -3.4e38f;
#pragma unroll
    for (int i = 0; i < 4; ++i) {
        int kidx = tid + i * 128;
        float sv = Sc[base + kidx] * 0.125f;
        bool bad = (mask[b * S_ + kidx] == 0) || (causal && kidx > q);
        sv = bad ? -1.0e9f : sv;
        v[i] = sv;
        mx = fmaxf(mx, sv);
    }
    __shared__ float red[128];
    red[tid] = mx; __syncthreads();
    for (int o = 64; o > 0; o >>= 1) { if (tid < o) red[tid] = fmaxf(red[tid], red[tid + o]); __syncthreads(); }
    mx = red[0]; __syncthreads();

    float s = 0.f;
#pragma unroll
    for (int i = 0; i < 4; ++i) { v[i] = __expf(v[i] - mx); s += v[i]; }
    red[tid] = s; __syncthreads();
    for (int o = 64; o > 0; o >>= 1) { if (tid < o) red[tid] += red[tid + o]; __syncthreads(); }
    float inv = 1.0f / red[0];
#pragma unroll
    for (int i = 0; i < 4; ++i)
        P[base + tid + i * 128] = (bf16)(v[i] * inv);
}

// y = LayerNorm(X + Y) * g + beta  -> f32 out + bf16 out ; one block per row (D=1024)
__global__ __launch_bounds__(256)
void k_add_ln(const float* __restrict__ X, const float* __restrict__ Y,
              const float* __restrict__ G, const float* __restrict__ Bt,
              float* __restrict__ Of, bf16* __restrict__ Ob)
{
    const int row = blockIdx.x;
    const int tid = threadIdx.x;
    const long long base = (long long)row * D_;

    float v[4]; float s = 0.f;
#pragma unroll
    for (int i = 0; i < 4; ++i) {
        int c = tid + i * 256;
        v[i] = X[base + c] + Y[base + c];
        s += v[i];
    }
    __shared__ float red[256];
    red[tid] = s; __syncthreads();
    for (int o = 128; o > 0; o >>= 1) { if (tid < o) red[tid] += red[tid + o]; __syncthreads(); }
    float mean = red[0] * (1.0f / D_);
    __syncthreads();

    float s2 = 0.f;
#pragma unroll
    for (int i = 0; i < 4; ++i) { float d = v[i] - mean; s2 += d * d; }
    red[tid] = s2; __syncthreads();
    for (int o = 128; o > 0; o >>= 1) { if (tid < o) red[tid] += red[tid + o]; __syncthreads(); }
    float rstd = rsqrtf(red[0] * (1.0f / D_) + 1e-5f);

#pragma unroll
    for (int i = 0; i < 4; ++i) {
        int c = tid + i * 256;
        float o = (v[i] - mean) * rstd * G[c] + Bt[c];
        Of[base + c] = o;
        Ob[base + c] = (bf16)o;
    }
}

// =====================================================================
// Host-side orchestration
// =====================================================================
namespace {

struct Scratch {
    float *Xe, *Xd, *qf, *kf, *vf, *Sc, *Obh, *Abuf, *T, *T2, *Fb, *Hid;
    bf16  *Xeb, *Xdb, *Qbh, *Kbh, *Vt, *P, *Mrg, *Tb, *T2b, *Hidb, *Wbf;
};

inline void gemm(hipStream_t st, float* C, const bf16* A, const bf16* W, const float* bias,
                 int M, int N, int K, int lda, int ldw, int ldc,
                 long long sA, long long sW, long long sC, int batch)
{
    dim3 blk(32, 4, 1), grd(N / 64, M / 128, batch);
    k_gemm<<<grd, blk, 0, st>>>(C, A, W, bias, K, lda, ldw, ldc, sA, sW, sC);
}

inline void to_bf16(hipStream_t st, const float* in, bf16* out, int n)
{
    k_f32_to_bf16<<<(n + 255) / 256, 256, 0, st>>>(in, out, n);
}

// full multi-head attention: out(f32, M_ x D_) = MHA(Xq, Xkv; w4 = [Wq,Wk,Wv,Wo])
void run_mha(hipStream_t st, const bf16* Xq, const bf16* Xkv, const float* w4,
             const int* mask, int causal, float* out, Scratch& s)
{
    const int nMD = M_ * D_;
    const long long DD = (long long)D_ * D_;

    // projections
    to_bf16(st, w4 + 0 * DD, s.Wbf, (int)DD);
    gemm(st, s.qf, Xq,  s.Wbf, nullptr, M_, D_, D_, D_, D_, D_, 0, 0, 0, 1);
    to_bf16(st, w4 + 1 * DD, s.Wbf, (int)DD);
    gemm(st, s.kf, Xkv, s.Wbf, nullptr, M_, D_, D_, D_, D_, D_, 0, 0, 0, 1);
    to_bf16(st, w4 + 2 * DD, s.Wbf, (int)DD);
    gemm(st, s.vf, Xkv, s.Wbf, nullptr, M_, D_, D_, D_, D_, D_, 0, 0, 0, 1);

    k_split_qk<<<(nMD + 255) / 256, 256, 0, st>>>(s.qf, s.Qbh);
    k_split_qk<<<(nMD + 255) / 256, 256, 0, st>>>(s.kf, s.Kbh);
    k_split_vt<<<(nMD + 255) / 256, 256, 0, st>>>(s.vf, s.Vt);

    // scores = Q @ K^T  (batch B*H, M=S, N=S, K=DH)
    gemm(st, s.Sc, s.Qbh, s.Kbh, nullptr, S_, S_, DH_, DH_, DH_, S_,
         (long long)S_ * DH_, (long long)S_ * DH_, (long long)S_ * S_, B_ * H_);

    k_softmax<<<B_ * H_ * S_, 128, 0, st>>>(s.Sc, s.P, mask, causal);

    // O = P @ V  via V^T  (batch B*H, M=S, N=DH, K=S)
    gemm(st, s.Obh, s.P, s.Vt, nullptr, S_, DH_, S_, S_, S_, DH_,
         (long long)S_ * S_, (long long)DH_ * S_, (long long)S_ * DH_, B_ * H_);

    k_merge_heads<<<(nMD + 255) / 256, 256, 0, st>>>(s.Obh, s.Mrg);

    to_bf16(st, w4 + 3 * DD, s.Wbf, (int)DD);
    gemm(st, out, s.Mrg, s.Wbf, nullptr, M_, D_, D_, D_, D_, D_, 0, 0, 0, 1);
}

void run_ffn(hipStream_t st, const bf16* Tin, const float* w1, const float* b1,
             const float* w2, const float* b2, float* out, Scratch& s)
{
    to_bf16(st, w1, s.Wbf, F_ * D_);
    gemm(st, s.Hid, Tin, s.Wbf, b1, M_, F_, D_, D_, D_, F_, 0, 0, 0, 1);
    k_relu_bf16<<<(M_ * F_ + 255) / 256, 256, 0, st>>>(s.Hid, s.Hidb, M_ * F_);
    to_bf16(st, w2, s.Wbf, D_ * F_);
    gemm(st, out, s.Hidb, s.Wbf, b2, M_, D_, F_, F_, F_, D_, 0, 0, 0, 1);
}

} // namespace

extern "C" void kernel_launch(void* const* d_in, const int* in_sizes, int n_in,
                              void* d_out, int out_size, void* d_ws, size_t ws_size,
                              hipStream_t stream)
{
    const float* src        = (const float*)d_in[0];
    const float* tgt        = (const float*)d_in[1];
    const float* enc_attn_w = (const float*)d_in[2];
    const float* enc_ln_g   = (const float*)d_in[3];
    const float* enc_ln_b   = (const float*)d_in[4];
    const float* enc_w1     = (const float*)d_in[5];
    const float* enc_b1     = (const float*)d_in[6];
    const float* enc_w2     = (const float*)d_in[7];
    const float* enc_b2     = (const float*)d_in[8];
    const float* dec_attn_w = (const float*)d_in[9];
    const float* dec_ln_g   = (const float*)d_in[10];
    const float* dec_ln_b   = (const float*)d_in[11];
    const float* dec_w1     = (const float*)d_in[12];
    const float* dec_b1     = (const float*)d_in[13];
    const float* dec_w2     = (const float*)d_in[14];
    const float* dec_b2     = (const float*)d_in[15];
    const int*   src_mask   = (const int*)d_in[16];
    const int*   tgt_mask   = (const int*)d_in[17];

    // ---- bump allocator over workspace ----
    char* wsp = (char*)d_ws;
    size_t off = 0;
    auto alloc = [&](size_t bytes) -> void* {
        void* p = wsp + off;
        off = (off + bytes + 255) & ~(size_t)255;
        return p;
    };

    const size_t fMD = (size_t)M_ * D_ * sizeof(float);
    const size_t bMD = (size_t)M_ * D_ * sizeof(bf16);

    Scratch s;
    s.Xe   = (float*)alloc(fMD);
    s.Xd   = (float*)alloc(fMD);
    s.qf   = (float*)alloc(fMD);
    s.kf   = (float*)alloc(fMD);
    s.vf   = (float*)alloc(fMD);
    s.Obh  = (float*)alloc(fMD);
    s.Abuf = (float*)alloc(fMD);
    s.T    = (float*)alloc(fMD);
    s.T2   = (float*)alloc(fMD);
    s.Fb   = (float*)alloc(fMD);
    s.Hid  = (float*)alloc((size_t)M_ * F_ * sizeof(float));
    s.Sc   = (float*)alloc((size_t)B_ * H_ * S_ * S_ * sizeof(float));
    s.Xeb  = (bf16*)alloc(bMD);
    s.Xdb  = (bf16*)alloc(bMD);
    s.Qbh  = (bf16*)alloc(bMD);
    s.Kbh  = (bf16*)alloc(bMD);
    s.Vt   = (bf16*)alloc(bMD);
    s.Mrg  = (bf16*)alloc(bMD);
    s.Tb   = (bf16*)alloc(bMD);
    s.T2b  = (bf16*)alloc(bMD);
    s.Hidb = (bf16*)alloc((size_t)M_ * F_ * sizeof(bf16));
    s.P    = (bf16*)alloc((size_t)B_ * H_ * S_ * S_ * sizeof(bf16));
    s.Wbf  = (bf16*)alloc((size_t)F_ * D_ * sizeof(bf16));

    const int nMD = M_ * D_;
    const long long DD = (long long)D_ * D_;

    // ================= encoder =================
    k_copy_both<<<(nMD + 255) / 256, 256, 0, stream>>>(src, s.Xe, s.Xeb, nMD);
    for (int i = 0; i < L_; ++i) {
        const float* w = enc_attn_w + (long long)i * 4 * DD;
        run_mha(stream, s.Xeb, s.Xeb, w, src_mask, 0, s.Abuf, s);
        k_add_ln<<<M_, 256, 0, stream>>>(s.Xe, s.Abuf,
                                         enc_ln_g + (i * 2 + 0) * D_,
                                         enc_ln_b + (i * 2 + 0) * D_, s.T, s.Tb);
        run_ffn(stream, s.Tb,
                enc_w1 + (long long)i * F_ * D_, enc_b1 + (long long)i * F_,
                enc_w2 + (long long)i * D_ * F_, enc_b2 + (long long)i * D_, s.Fb, s);
        k_add_ln<<<M_, 256, 0, stream>>>(s.T, s.Fb,
                                         enc_ln_g + (i * 2 + 1) * D_,
                                         enc_ln_b + (i * 2 + 1) * D_, s.Xe, s.Xeb);
    }

    // ================= decoder =================
    k_copy_both<<<(nMD + 255) / 256, 256, 0, stream>>>(tgt, s.Xd, s.Xdb, nMD);
    for (int i = 0; i < L_; ++i) {
        const float* w = dec_attn_w + (long long)i * 8 * DD;
        // masked self-attention on decoder layer input
        run_mha(stream, s.Xdb, s.Xdb, w, tgt_mask, 1, s.Abuf, s);
        k_add_ln<<<M_, 256, 0, stream>>>(s.Xd, s.Abuf,
                                         dec_ln_g + (i * 3 + 0) * D_,
                                         dec_ln_b + (i * 3 + 0) * D_, s.T, s.Tb);
        // cross-attention: Q from encoder output, K/V from raw decoder layer input
        run_mha(stream, s.Xeb, s.Xdb, w + 4 * DD, tgt_mask, 0, s.Abuf, s);
        k_add_ln<<<M_, 256, 0, stream>>>(s.T, s.Abuf,
                                         dec_ln_g + (i * 3 + 1) * D_,
                                         dec_ln_b + (i * 3 + 1) * D_, s.T2, s.T2b);
        run_ffn(stream, s.T2b,
                dec_w1 + (long long)i * F_ * D_, dec_b1 + (long long)i * F_,
                dec_w2 + (long long)i * D_ * F_, dec_b2 + (long long)i * D_, s.Fb, s);
        k_add_ln<<<M_, 256, 0, stream>>>(s.T2, s.Fb,
                                         dec_ln_g + (i * 3 + 2) * D_,
                                         dec_ln_b + (i * 3 + 2) * D_, s.Xd, s.Xdb);
    }

    hipMemcpyAsync(d_out, s.Xd, fMD, hipMemcpyDeviceToDevice, stream);
}
